// Mukara_14061722927420
// MI455X (gfx1250) — matrix operand order repaired and checked by reading, not compiled, not run on hardware
//
#include <hip/hip_runtime.h>
#include <hip/hip_bf16.h>

// ---------------------------------------------------------------------------
// Types for WMMA fragments (CDNA5 / gfx1250, wave32)
// ---------------------------------------------------------------------------
typedef _Float16 v16h __attribute__((ext_vector_type(16)));
typedef _Float16 h8   __attribute__((ext_vector_type(8)));
typedef _Float16 h4   __attribute__((ext_vector_type(4)));
typedef float    v8f  __attribute__((ext_vector_type(8)));

union Frag16 { v16h v; h8 h[2]; };

// ---------------------------------------------------------------------------
// Templated fused GEMM:  C[M,NN] = act( [A0|A1][M,KA] @ W[KA,NN] + bias )
//  - KA0/KA1/NN compile-time: K loop fully unrolled, shifts not div/mod,
//    immediate-offset stores, exact LDS sizing per instantiation
//  - 64 rows per block, 256 threads (8 wave32 waves)
//  - full K staged in LDS as fp16; W stored transposed [NN][KAPAD] so both
//    A and B fragments are two contiguous ds_load_b128 per lane
//  - v_wmma_f32_16x16x32_f16, fp32 accumulate, fused bias + relu on store
// Fragment layout per CDNA5 ISA (16-bit A 16x32 / B 32x16 / f32 D):
//  A: lane l elems 0..7  = A[M=l&15][k0+(l>>4)*8 .. +7], elems 8..15 at +16
//  B: lane l elems 0..7  = B[k0+(l>>4)*8 .. +7][N=l&15], elems 8..15 at +16
//  D: lane l VGPR v: row = v + (l>>4)*8, col = l&15
// ---------------------------------------------------------------------------
template<int KA0, int KA1, int NN, bool RELU, bool HASBIAS>
__global__ __launch_bounds__(256)
void gemm_wmma(const float* __restrict__ A0, const float* __restrict__ A1,
               const float* __restrict__ W,  const float* __restrict__ bias,
               float* __restrict__ C, int M)
{
    constexpr int KA     = KA0 + KA1;
    constexpr int KAEFF  = (KA + 31) & ~31;   // pad K to WMMA step (power of 2 here)
    constexpr int KAPAD  = KAEFF + 8;         // +16B to dodge LDS bank conflicts
    constexpr int TILESN = NN / 16;

    __shared__ __align__(16) _Float16 sW[NN * KAPAD];  // W^T [NN][KAPAD]
    __shared__ __align__(16) _Float16 sA[64 * KAPAD];  // A   [64][KAPAD]

    const int row0 = blockIdx.x * 64;
    const bool fullBlock = (row0 + 64 <= M);

    // ---- stage W^T: linear (coalesced) read of W[idx = k*NN + n] ----
    for (int idx = threadIdx.x; idx < NN * KAEFF; idx += 256) {
        int k = idx / NN;          // compile-time power-of-2 -> shift
        int n = idx % NN;
        _Float16 v = (_Float16)0.0f;
        if (k < KA) v = (_Float16)W[idx];
        sW[n * KAPAD + k] = v;
    }
    // ---- stage A rows: float4 loads (coalesced over K), concat A0|A1 ----
    // KA, KA0 are multiples of 4 so a float4 group never straddles a boundary
    for (int idx = threadIdx.x; idx < 64 * KAEFF / 4; idx += 256) {
        int r = (idx * 4) / KAEFF;
        int k = (idx * 4) % KAEFF;
        int row = row0 + r;
        h4 o = { (_Float16)0.f, (_Float16)0.f, (_Float16)0.f, (_Float16)0.f };
        if ((fullBlock || row < M) && k < KA) {
            const float* sp = (k < KA0) ? (A0 + (size_t)row * KA0 + k)
                                        : (A1 + (size_t)row * KA1 + (k - KA0));
            float4 f = *(const float4*)sp;
            o[0] = (_Float16)f.x; o[1] = (_Float16)f.y;
            o[2] = (_Float16)f.z; o[3] = (_Float16)f.w;
        }
        *(h4*)&sA[r * KAPAD + k] = o;   // 8B aligned: KAPAD%8==0, k%4==0
    }
    __syncthreads();

    const int wave = threadIdx.x >> 5;
    const int lane = threadIdx.x & 31;
    const int l15  = lane & 15;
    const int lhi  = lane >> 4;

    #pragma unroll
    for (int t0 = 0; t0 < 4 * TILESN; t0 += 8) {
        const int t  = t0 + wave;
        const int mt = t & 3;
        const int nt = t >> 2;

        v8f acc = {0.f, 0.f, 0.f, 0.f, 0.f, 0.f, 0.f, 0.f};

        const _Float16* aBase = &sA[(mt * 16 + l15) * KAPAD + lhi * 8];
        const _Float16* bBase = &sW[(nt * 16 + l15) * KAPAD + lhi * 8];

        #pragma unroll
        for (int k0 = 0; k0 < KAEFF; k0 += 32) {
            Frag16 a, b;
            a.h[0] = *(const h8*)(aBase + k0);        // K = k0 .. +7 (lane-split)
            a.h[1] = *(const h8*)(aBase + k0 + 16);   // K = k0+16 .. (lane-split)
            b.h[0] = *(const h8*)(bBase + k0);        // K = k0 .. +7 (lane-split)
            b.h[1] = *(const h8*)(bBase + k0 + 16);   // K = k0+16 .. (lane-split)
            acc = __builtin_amdgcn_wmma_f32_16x16x32_f16(
                false, a.v, false, b.v, (short)0, acc, false, false);
        }

        const int col  = nt * 16 + l15;
        const int rloc = mt * 16 + (lhi << 3);
        float bval = HASBIAS ? bias[col] : 0.f;
        float* cp = C + (size_t)(row0 + rloc) * NN + col;

        if (fullBlock) {                       // 4687/4688 blocks: branchless
            #pragma unroll
            for (int v = 0; v < 8; ++v) {
                float val = acc[v] + bval;
                if (RELU) val = fmaxf(val, 0.f);
                cp[v * NN] = val;              // immediate offsets, one base
            }
        } else {
            #pragma unroll
            for (int v = 0; v < 8; ++v) {
                if (row0 + rloc + v < M) {
                    float val = acc[v] + bval;
                    if (RELU) val = fmaxf(val, 0.f);
                    cp[v * NN] = val;
                }
            }
        }
    }
}

// ---------------------------------------------------------------------------
// Utility: fill buffer with constant
// ---------------------------------------------------------------------------
__global__ void fill_kernel(float* __restrict__ p, long n, float v) {
    long i = (long)blockIdx.x * blockDim.x + threadIdx.x;
    long stride = (long)gridDim.x * blockDim.x;
    for (; i < n; i += stride) p[i] = v;
}

// float atomic max via sign-split int/uint atomics (IEEE order preserving)
__device__ __forceinline__ void atomicMaxFloat(float* addr, float v) {
    if (v >= 0.f) atomicMax((int*)addr, __float_as_int(v));
    else          atomicMin((unsigned int*)addr, __float_as_uint(v));
}

// ---------------------------------------------------------------------------
// Edge pass 1: per-edge, per-head attention logits + segment max into m[dst]
// score[e,h] = 0.25 * dot(Q[src[e]][h*16:], K[dst[e]][h*16:])
// ---------------------------------------------------------------------------
__global__ void score_kernel(const float* __restrict__ Q, const float* __restrict__ K,
                             const int* __restrict__ src, const int* __restrict__ dst,
                             float* __restrict__ score, float* __restrict__ m, int E)
{
    int e = blockIdx.x * blockDim.x + threadIdx.x;
    if (e >= E) return;
    int s = src[e], d = dst[e];
    const float4* q4 = (const float4*)(Q + (size_t)s * 64);
    const float4* k4 = (const float4*)(K + (size_t)d * 64);
    #pragma unroll
    for (int hh = 0; hh < 4; ++hh) {
        float acc = 0.f;
        #pragma unroll
        for (int j = 0; j < 4; ++j) {
            float4 a = q4[hh * 4 + j];
            float4 b = k4[hh * 4 + j];
            acc += a.x * b.x + a.y * b.y + a.z * b.z + a.w * b.w;
        }
        acc *= 0.25f;   // 1/sqrt(HD=16)
        score[(size_t)e * 4 + hh] = acc;
        atomicMaxFloat(&m[(size_t)d * 4 + hh], acc);
    }
}

// ---------------------------------------------------------------------------
// Edge pass 2: e = exp(score - m[dst]); denom[dst] += e  (in-place score)
// ---------------------------------------------------------------------------
__global__ void expsum_kernel(const int* __restrict__ dst,
                              float* __restrict__ score,
                              const float* __restrict__ m,
                              float* __restrict__ denom, int E)
{
    int e = blockIdx.x * blockDim.x + threadIdx.x;
    if (e >= E) return;
    int d = dst[e];
    #pragma unroll
    for (int hh = 0; hh < 4; ++hh) {
        float ex = __expf(score[(size_t)e * 4 + hh] - m[(size_t)d * 4 + hh]);
        score[(size_t)e * 4 + hh] = ex;
        atomicAdd(&denom[(size_t)d * 4 + hh], ex);
    }
}

// ---------------------------------------------------------------------------
// Edge pass 3: agg[dst] += (e/denom[dst]) * V[src]   (one thread per edge*head)
// ---------------------------------------------------------------------------
__global__ void agg_kernel(const float* __restrict__ V,
                           const int* __restrict__ src, const int* __restrict__ dst,
                           const float* __restrict__ score, const float* __restrict__ denom,
                           float* __restrict__ agg, int E)
{
    int t = blockIdx.x * blockDim.x + threadIdx.x;
    if (t >= E * 4) return;
    int e = t >> 2, hh = t & 3;
    int s = src[e], d = dst[e];
    float attn = score[(size_t)e * 4 + hh] / denom[(size_t)d * 4 + hh];
    const float4* vp = (const float4*)(V + (size_t)s * 64 + hh * 16);
    float* ap = agg + (size_t)d * 64 + hh * 16;
    #pragma unroll
    for (int j = 0; j < 4; ++j) {
        float4 v = vp[j];
        atomicAdd(&ap[j * 4 + 0], attn * v.x);
        atomicAdd(&ap[j * 4 + 1], attn * v.y);
        atomicAdd(&ap[j * 4 + 2], attn * v.z);
        atomicAdd(&ap[j * 4 + 3], attn * v.w);
    }
}

// ---------------------------------------------------------------------------
// Residual + LayerNorm over D=64, one wave32 per node (2 elems/lane)
// h = LN(h + hn) * g + b
// ---------------------------------------------------------------------------
__global__ __launch_bounds__(256)
void ln_kernel(float* __restrict__ h, const float* __restrict__ hn,
               const float* __restrict__ g, const float* __restrict__ b, int N)
{
    int wave = threadIdx.x >> 5, lane = threadIdx.x & 31;
    int n = blockIdx.x * 8 + wave;
    if (n >= N) return;
    float2 a = ((const float2*)h)[(size_t)n * 32 + lane];
    float2 c = ((const float2*)hn)[(size_t)n * 32 + lane];
    float x0 = a.x + c.x, x1 = a.y + c.y;
    float s = x0 + x1, ss = x0 * x0 + x1 * x1;
    #pragma unroll
    for (int off = 16; off > 0; off >>= 1) {
        s  += __shfl_xor(s, off);
        ss += __shfl_xor(ss, off);
    }
    float mean = s * (1.f / 64.f);
    float var  = ss * (1.f / 64.f) - mean * mean;
    float inv  = rsqrtf(var + 1e-5f);
    int d0 = lane * 2;
    float y0 = g[d0]     * (x0 - mean) * inv + b[d0];
    float y1 = g[d0 + 1] * (x1 - mean) * inv + b[d0 + 1];
    ((float2*)h)[(size_t)n * 32 + lane] = make_float2(y0, y1);
}

// ---------------------------------------------------------------------------
// Final 128 -> 1 projection: one wave32 per node, float4 per lane + reduce
// ---------------------------------------------------------------------------
__global__ __launch_bounds__(256)
void outvec_kernel(const float* __restrict__ t1, const float* __restrict__ w2,
                   const float* __restrict__ b2, float* __restrict__ out, int N)
{
    int wave = threadIdx.x >> 5, lane = threadIdx.x & 31;
    int n = blockIdx.x * 8 + wave;
    if (n >= N) return;
    float4 a = ((const float4*)(t1 + (size_t)n * 128))[lane];
    float4 w = ((const float4*)w2)[lane];
    float s = a.x * w.x + a.y * w.y + a.z * w.z + a.w * w.w;
    #pragma unroll
    for (int off = 16; off > 0; off >>= 1) s += __shfl_xor(s, off);
    if (lane == 0) out[n] = s + b2[0];
}

// ---------------------------------------------------------------------------
// Host launch
// ---------------------------------------------------------------------------
extern "C" void kernel_launch(void* const* d_in, const int* in_sizes, int n_in,
                              void* d_out, int out_size, void* d_ws, size_t ws_size,
                              hipStream_t stream)
{
    const int F = 16, D = 64, HID = 128, L = 3;
    const int N = in_sizes[0] / F;
    const int E = in_sizes[1];

    const float* ef     = (const float*)d_in[0];
    const int*   src    = (const int*)  d_in[1];
    const int*   dst    = (const int*)  d_in[2];
    const float* ep_w1  = (const float*)d_in[3];
    const float* ep_b1  = (const float*)d_in[4];
    const float* ep_w2  = (const float*)d_in[5];
    const float* ep_b2  = (const float*)d_in[6];
    const float* wq     = (const float*)d_in[7];
    const float* wk     = (const float*)d_in[8];
    const float* wv     = (const float*)d_in[9];
    const float* up_w1  = (const float*)d_in[10];
    const float* up_b1  = (const float*)d_in[11];
    const float* up_w2  = (const float*)d_in[12];
    const float* up_b2  = (const float*)d_in[13];
    const float* ln_g   = (const float*)d_in[14];
    const float* ln_b   = (const float*)d_in[15];
    const float* out_w1 = (const float*)d_in[16];
    const float* out_b1 = (const float*)d_in[17];
    const float* out_w2 = (const float*)d_in[18];
    const float* out_b2 = (const float*)d_in[19];
    float* out = (float*)d_out;

    // workspace carve (256B aligned regions)
    auto align256 = [](size_t x) { return (x + 255) & ~(size_t)255; };
    char* w = (char*)d_ws;
    size_t off = 0;
    auto take = [&](size_t bytes) { float* p = (float*)(w + off); off += align256(bytes); return p; };
    float* tmp1  = take((size_t)N * 128 * 4);  // MLP hidden activations
    float* h     = take((size_t)N * 64 * 4);   // node embeddings
    float* Qb    = take((size_t)N * 64 * 4);   // Q; recycled as h_new
    float* Kb    = take((size_t)N * 64 * 4);
    float* Vb    = take((size_t)N * 64 * 4);
    float* aggb  = take((size_t)N * 64 * 4);
    float* score = take((size_t)E * 4 * 4);
    float* mbuf  = take((size_t)N * 4 * 4);
    float* dbuf  = take((size_t)N * 4 * 4);
    (void)ws_size; (void)n_in; (void)out_size;

    const int gemmBlocks = (N + 63) / 64;
    const int edgeBlocks = (E + 255) / 256;
    const int ehBlocks   = (E * 4 + 255) / 256;
    const int nodeBlocks = (N + 7) / 8;
    auto fillBlocks = [](long n) { long b = (n + 255) / 256; return (int)(b > 65535 ? 65535 : b); };

    // ---- edge feature projection: h = relu(ef@W1+b1)@W2+b2 ----
    gemm_wmma<16, 0, 128, true, true><<<gemmBlocks, 256, 0, stream>>>(ef, nullptr, ep_w1, ep_b1, tmp1, N);
    gemm_wmma<128, 0, 64, false, true><<<gemmBlocks, 256, 0, stream>>>(tmp1, nullptr, ep_w2, ep_b2, h, N);

    // ---- GNN layers ----
    for (int i = 0; i < L; ++i) {
        gemm_wmma<64, 0, 64, false, false><<<gemmBlocks, 256, 0, stream>>>(h, nullptr, wq + (size_t)i * D * D, nullptr, Qb, N);
        gemm_wmma<64, 0, 64, false, false><<<gemmBlocks, 256, 0, stream>>>(h, nullptr, wk + (size_t)i * D * D, nullptr, Kb, N);
        gemm_wmma<64, 0, 64, false, false><<<gemmBlocks, 256, 0, stream>>>(h, nullptr, wv + (size_t)i * D * D, nullptr, Vb, N);

        fill_kernel<<<fillBlocks((long)N * 4), 256, 0, stream>>>(mbuf, (long)N * 4, -1e30f);
        fill_kernel<<<fillBlocks((long)N * 4), 256, 0, stream>>>(dbuf, (long)N * 4, 0.f);
        fill_kernel<<<fillBlocks((long)N * 64), 256, 0, stream>>>(aggb, (long)N * 64, 0.f);

        score_kernel <<<edgeBlocks, 256, 0, stream>>>(Qb, Kb, src, dst, score, mbuf, E);
        expsum_kernel<<<edgeBlocks, 256, 0, stream>>>(dst, score, mbuf, dbuf, E);
        agg_kernel   <<<ehBlocks,   256, 0, stream>>>(Vb, src, dst, score, dbuf, aggb, E);

        // update_net on concat [h | agg] (K split handled inside the GEMM)
        gemm_wmma<64, 64, 128, true, true><<<gemmBlocks, 256, 0, stream>>>(
            h, aggb, up_w1 + (size_t)i * 2 * D * HID, up_b1 + (size_t)i * HID, tmp1, N);
        gemm_wmma<128, 0, 64, false, true><<<gemmBlocks, 256, 0, stream>>>(
            tmp1, nullptr, up_w2 + (size_t)i * HID * D, up_b2 + (size_t)i * D, Qb, N); // h_new -> Qb
        ln_kernel<<<nodeBlocks, 256, 0, stream>>>(h, Qb, ln_g + (size_t)i * D, ln_b + (size_t)i * D, N);
    }

    // ---- output MLP ----
    gemm_wmma<64, 0, 128, true, true><<<gemmBlocks, 256, 0, stream>>>(h, nullptr, out_w1, out_b1, tmp1, N);
    outvec_kernel<<<nodeBlocks, 256, 0, stream>>>(tmp1, out_w2, out_b2, out, N);
}
